// OCCModel_16209206575262
// MI455X (gfx1250) — matrix operand order repaired and checked by reading, not compiled, not run on hardware
//
#include <hip/hip_runtime.h>
#include <cmath>

typedef __attribute__((ext_vector_type(16))) _Float16 v16h;
typedef __attribute__((ext_vector_type(8)))  _Float16 v8h;
typedef __attribute__((ext_vector_type(8)))  float    v8f;

#define ROWSTRIDE 136   // halves per activation row: 128 + 8 pad (272B -> bank stagger)
#define OFF_L0   0      // offsets in halves inside d_ws fragment buffer
#define OFF_L1   8192
#define OFF_L2   24576
#define OFF_L3   40960
#define OFF_OUT  57344
#define FRAG_TOTAL 59392   // 118,784 bytes of f16 weight fragments in d_ws

// ---------------------------------------------------------------------------
// Prep: convert f32 weights into per-lane WMMA B-fragments (f16) in d_ws.
// B-matrix (KxN) layout for V_WMMA_F32_16X16X32_F16 (ISA 7.12.2):
//   lane holds column N = nb*16 + (lane&15)
//   halves hold K in A-table order: h<8 -> K=kb*32+8*(lane>>4)+h, h>=8 -> K=kb*32+16+8*(lane>>4)+(h-8)
// ---------------------------------------------------------------------------
__global__ void occ_prep_weights(const float* __restrict__ W0,
                                 const float* __restrict__ W1,
                                 const float* __restrict__ W2,
                                 const float* __restrict__ W3,
                                 const float* __restrict__ Wout,
                                 _Float16* __restrict__ frag) {
  int t = blockIdx.x * blockDim.x + threadIdx.x;
  if (t >= FRAG_TOTAL) return;
  int base, kbn, kact, nact;
  const float* W;
  if (t < OFF_L1)       { base = OFF_L0;  kbn = 2; kact = 38;  nact = 128; W = W0;   }
  else if (t < OFF_L2)  { base = OFF_L1;  kbn = 4; kact = 128; nact = 128; W = W1;   }
  else if (t < OFF_L3)  { base = OFF_L2;  kbn = 4; kact = 128; nact = 128; W = W2;   }
  else if (t < OFF_OUT) { base = OFF_L3;  kbn = 4; kact = 128; nact = 128; W = W3;   }
  else                  { base = OFF_OUT; kbn = 4; kact = 128; nact = 1;   W = Wout; }
  int r    = t - base;
  int h    = r & 15;
  int lane = (r >> 4) & 31;
  int kb   = (r >> 9) % kbn;
  int nb   = (r >> 9) / kbn;
  int k = kb * 32 + ((lane >> 4) << 3) + (h & 7) + ((h >> 3) << 4);
  int n = nb * 16 + (lane & 15);
  float v = 0.0f;
  if (k < kact && n < nact) v = W[k * nact + n];
  frag[t] = (_Float16)v;
}

// ---------------------------------------------------------------------------
// A-matrix (16x32, MxK) fragment load from row-major f16 LDS row.
// ISA layout: lane (0-15)=M, VGPR0..3 halves = K klo..klo+7, VGPR4..7 = K 16+klo..,
// klo = 0 (lanes 0-15) / 8 (lanes 16-31)  => two contiguous 16-byte LDS loads.
// ---------------------------------------------------------------------------
__device__ inline v16h afrag(const _Float16* rowp, int kblk, int klo) {
  union { v16h v; v8h h[2]; } u;
  u.h[0] = *(const v8h*)(rowp + kblk * 32 + klo);
  u.h[1] = *(const v8h*)(rowp + kblk * 32 + 16 + klo);
  return u.v;
}

template<int KB>
__device__ inline void load_afrags(v16h* A, const _Float16* r0, const _Float16* r1, int klo) {
  #pragma unroll
  for (int kb = 0; kb < KB; ++kb) {
    A[kb]      = afrag(r0, kb, klo);   // M-tile 0 (points 0-15)
    A[KB + kb] = afrag(r1, kb, klo);   // M-tile 1 (points 16-31)
  }
}

// Single-instruction ReLU: v_med3_f32(x, 0, +inf) == max(x,0), emitted verbatim
// (no sNaN canonicalize pre-op, unlike fmaxf/maxnum which cost 2 VALU each).
__device__ inline float relu1(float u) {
  return __builtin_amdgcn_fmed3f(u, 0.0f, __builtin_inff());
}

// One 128-wide dense layer for 32 points: D(32x128) = A(32,KB*32) x W + bias [, ReLU]
// Bias is injected as the C operand of the first WMMA in each K-chain (free add).
// unroll-2 on the N-block loop -> two independent WMMA chains interleave, filling
// the WMMA->VALU RAW hazard slots with useful work.
template<int KB, bool RELU>
__device__ inline void run_dense(const v16h* A, const _Float16* __restrict__ wf,
                                 const float* __restrict__ bias,
                                 _Float16* stagw, int lane) {
  const int nrow = lane & 15;
  const int mhi  = (lane >> 4) << 3;   // C/D layout: lanes 16-31 hold rows M+8

  #pragma unroll 2
  for (int nb = 0; nb < 8; ++nb) {
    float bv = bias[nb * 16 + nrow];   // per-lane bias, pipelined global_load_b32
    v8f acc0, acc1;
    #pragma unroll
    for (int r = 0; r < 8; ++r) { acc0[r] = bv; acc1[r] = bv; }
    #pragma unroll
    for (int kb = 0; kb < KB; ++kb) {
      v16h bf = *(const v16h*)(wf + ((nb * KB + kb) * 32 + lane) * 16);
      acc0 = __builtin_amdgcn_wmma_f32_16x16x32_f16(false, A[kb],      false, bf,
                                                    (short)0, acc0, false, false);
      acc1 = __builtin_amdgcn_wmma_f32_16x16x32_f16(false, A[KB + kb], false, bf,
                                                    (short)0, acc1, false, false);
    }
    #pragma unroll
    for (int r2 = 0; r2 < 8; ++r2) {
      float u0 = acc0[r2];
      float u1 = acc1[r2];
      if (RELU) { u0 = relu1(u0); u1 = relu1(u1); }
      stagw[(mhi + r2) * ROWSTRIDE + nb * 16 + nrow]      = (_Float16)u0;
      stagw[(16 + mhi + r2) * ROWSTRIDE + nb * 16 + nrow] = (_Float16)u1;
    }
  }
}

// Warm the WGP$ with the next layer's B-fragments (global_prefetch_b8).
__device__ inline void prefetch_frags(const _Float16* p, int lane) {
  __builtin_prefetch((const void*)(p + (size_t)lane * 1024), 0, 1);
}

// ---------------------------------------------------------------------------
// Main kernel: 256 threads = 8 waves, each wave fully owns 32 points.
// No barriers: per-wave LDS staging slice, DS ops are in-order per wave.
// ---------------------------------------------------------------------------
__global__ __launch_bounds__(256) void occ_mlp_kernel(
    const float* __restrict__ x, const float* __restrict__ xz,
    const float* __restrict__ aabb,
    const float* __restrict__ b0, const float* __restrict__ b1,
    const float* __restrict__ b2, const float* __restrict__ b3,
    const float* __restrict__ bout, const int* __restrict__ nvp,
    const _Float16* __restrict__ frag, float* __restrict__ out)
{
  extern __shared__ _Float16 smem[];           // 8 waves x 32 rows x ROWSTRIDE halves
  const int tid  = threadIdx.x;
  const int w    = tid >> 5;
  const int lane = tid & 31;
  _Float16* stagw = smem + w * 32 * ROWSTRIDE;

  const int p0 = (blockIdx.x * 8 + w) * 32;    // first point of this wave
  const int p  = p0 + lane;                    // this lane's point (32 lanes = 32 pts)
  const int nv = *nvp;
  const float nvf = (float)nv;

  prefetch_frags(frag + OFF_L0, lane);

  // ---- gather: voxel feature + positional encode, write f16 row to LDS ----
  const int b = p >> 16;                       // NPTS = 65536
  float px = x[p * 3 + 0], py = x[p * 3 + 1], pz = x[p * 3 + 2];
  float lox = aabb[0], loy = aabb[1], loz = aabb[2];
  float hix = aabb[3], hiy = aabb[4], hiz = aabb[5];
  float icx = (px - lox) / (hix - lox) * nvf;
  float icy = (py - loy) / (hiy - loy) * nvf;
  float icz = (pz - loz) / (hiz - loz) * nvf;
  float vx0 = fminf(fmaxf(floorf(icx), 0.0f), nvf - 1.0f);
  float vx1 = fminf(fmaxf(floorf(icy), 0.0f), nvf - 1.0f);
  float vx2 = fminf(fmaxf(floorf(icz), 0.0f), nvf - 1.0f);
  int flat = (((int)vx0) * nv + (int)vx1) * nv + (int)vx2;

  const float4* f4 = (const float4*)(xz + ((size_t)b * (size_t)(nv * nv * nv) + flat) * 32);
  _Float16* row = stagw + lane * ROWSTRIDE;
  #pragma unroll
  for (int j = 0; j < 8; ++j) {
    float4 f = f4[j];
    row[j * 4 + 0] = (_Float16)f.x;
    row[j * 4 + 1] = (_Float16)f.y;
    row[j * 4 + 2] = (_Float16)f.z;
    row[j * 4 + 3] = (_Float16)f.w;
  }
  row[32] = (_Float16)((icx - vx0) * 2.0f - 1.0f);
  row[33] = (_Float16)((icy - vx1) * 2.0f - 1.0f);
  row[34] = (_Float16)((icz - vx2) * 2.0f - 1.0f);
  row[35] = (_Float16)px;
  row[36] = (_Float16)py;
  row[37] = (_Float16)pz;
  #pragma unroll
  for (int j = 38; j < 64; ++j) row[j] = (_Float16)0.0f;   // pad K 38->64

  const int mrow = lane & 15;
  const int klo  = (lane >> 4) << 3;           // A-layout half offset: 0 or 8
  const _Float16* r0 = stagw + mrow * ROWSTRIDE;
  const _Float16* r1 = stagw + (16 + mrow) * ROWSTRIDE;

  // layer 0: (32x64 padded) x (64x128)
  v16h A0[4];
  load_afrags<2>(A0, r0, r1, klo);
  prefetch_frags(frag + OFF_L1, lane);
  run_dense<2, true>(A0, frag + OFF_L0, b0, stagw, lane);

  // layer 1
  v16h A1[8];
  load_afrags<4>(A1, r0, r1, klo);
  prefetch_frags(frag + OFF_L2, lane);
  run_dense<4, true>(A1, frag + OFF_L1, b1, stagw, lane);

  // layer 2 — its input IS the skip tensor; keep fragments in registers
  v16h S[8];
  load_afrags<4>(S, r0, r1, klo);
  prefetch_frags(frag + OFF_L3, lane);
  run_dense<4, true>(S, frag + OFF_L2, b2, stagw, lane);

  // layer 3
  v16h A3[8];
  load_afrags<4>(A3, r0, r1, klo);
  prefetch_frags(frag + OFF_OUT, lane);
  run_dense<4, true>(A3, frag + OFF_L3, b3, stagw, lane);

  // head: (relu(z3) + skip) @ Wout (N padded to 16), tanh
  v16h AF[8];
  load_afrags<4>(AF, r0, r1, klo);
  #pragma unroll
  for (int i = 0; i < 8; ++i) AF[i] = AF[i] + S[i];   // skip add in A-layout

  float bo = bout[0];
  v8f acc0, acc1;
  #pragma unroll
  for (int r = 0; r < 8; ++r) { acc0[r] = bo; acc1[r] = bo; }
  #pragma unroll
  for (int kb = 0; kb < 4; ++kb) {
    v16h bf = *(const v16h*)(frag + OFF_OUT + (kb * 32 + lane) * 16);
    acc0 = __builtin_amdgcn_wmma_f32_16x16x32_f16(false, AF[kb],     false, bf,
                                                  (short)0, acc0, false, false);
    acc1 = __builtin_amdgcn_wmma_f32_16x16x32_f16(false, AF[4 + kb], false, bf,
                                                  (short)0, acc1, false, false);
  }
  if ((lane & 15) == 0) {                      // only column N=0 is the real output
    int mb = (lane >> 4) * 8;
    #pragma unroll
    for (int r2 = 0; r2 < 8; ++r2) {
      out[p0 + mb + r2]      = tanhf(acc0[r2]);
      out[p0 + 16 + mb + r2] = tanhf(acc1[r2]);
    }
  }
}

extern "C" void kernel_launch(void* const* d_in, const int* in_sizes, int n_in,
                              void* d_out, int out_size, void* d_ws, size_t ws_size,
                              hipStream_t stream) {
  const float* x    = (const float*)d_in[0];
  const float* xz   = (const float*)d_in[1];
  const float* aabb = (const float*)d_in[2];
  const float* W0   = (const float*)d_in[3];
  const float* b0   = (const float*)d_in[4];
  const float* W1   = (const float*)d_in[5];
  const float* b1   = (const float*)d_in[6];
  const float* W2   = (const float*)d_in[7];
  const float* b2   = (const float*)d_in[8];
  const float* W3   = (const float*)d_in[9];
  const float* b3   = (const float*)d_in[10];
  const float* Wout = (const float*)d_in[11];
  const float* bout = (const float*)d_in[12];
  const int*   nvp  = (const int*)d_in[13];
  _Float16* frag = (_Float16*)d_ws;
  float* out = (float*)d_out;

  occ_prep_weights<<<(FRAG_TOTAL + 255) / 256, 256, 0, stream>>>(W0, W1, W2, W3, Wout, frag);

  const int n_points = in_sizes[0] / 3;                 // B*NPTS = 524288
  const int blocks   = n_points / 256;                  // 256 points per block
  const size_t lds   = (size_t)8 * 32 * ROWSTRIDE * sizeof(_Float16);  // 69,632 B
  occ_mlp_kernel<<<blocks, 256, lds, stream>>>(x, xz, aabb, b0, b1, b2, b3,
                                               bout, nvp, frag, out);
}
